// TransformerLayer_89824946028698
// MI455X (gfx1250) — compile-verified, matrix operand order checked by reading, not run on hardware
//
#include <hip/hip_runtime.h>

// ---------------------------------------------------------------------------
// Transformer layer for MI455X (gfx1250): bf16 WMMA GEMMs + flash attention.
// T=1024, B=4, D=1024, H=16, HD=64, M = T*B = 4096.
// ---------------------------------------------------------------------------

typedef __attribute__((ext_vector_type(16))) __bf16 v16bf;
typedef __attribute__((ext_vector_type(8)))  float  v8f;

#define TT   1024
#define BB   4
#define DD   1024
#define HH   16
#define HD   64
#define MM   (TT * BB)   // 4096

// CDNA5 async global->LDS path (ASYNCcnt-tracked) -- builtins probe-confirmed
// to exist here; parameter type is int __vector(4) with address spaces 1/3.
#define TRY_ASYNC 1
#if TRY_ASYNC && defined(__has_builtin)
# if __has_builtin(__builtin_amdgcn_global_load_async_to_lds_b128) && \
     __has_builtin(__builtin_amdgcn_s_wait_asynccnt)
#  define USE_ASYNC_LDS 1
# endif
#endif
#ifndef USE_ASYNC_LDS
# define USE_ASYNC_LDS 0
#endif

union FragU {
    uint4          u[2];
    unsigned short s[16];
    v16bf          v;
};

__device__ __forceinline__ unsigned short f2bf(float f) {
    union { float f; unsigned int u; } x; x.f = f;
    unsigned int u = x.u;
    u += 0x7FFFu + ((u >> 16) & 1u);     // round-to-nearest-even
    return (unsigned short)(u >> 16);
}

// 16-bit A/B fragment (16x32 MxK, or 32x16 KxN with lane==N) load.
// ISA layout: lanes 0-15 hold K {0..7,16..23}, lanes 16-31 hold {8..15,24..31}.
__device__ __forceinline__ v16bf frag_from_lds(const unsigned short* tile,
                                               int rowBase, int stride) {
    int lane = threadIdx.x & 31;
    int r    = lane & 15;
    int half = (lane >> 4) << 3;          // 0 or 8
    FragU f;
    const unsigned short* p = tile + (size_t)(rowBase + r) * stride + half;
    f.u[0] = *(const uint4*)(p);
    f.u[1] = *(const uint4*)(p + 16);
    return f.v;
}

__device__ __forceinline__ v16bf frag_from_global(const unsigned short* p0,
                                                  int rowStride) {
    int lane = threadIdx.x & 31;
    int r    = lane & 15;
    int half = (lane >> 4) << 3;
    FragU f;
    const unsigned short* p = p0 + (size_t)r * rowStride + half;
    f.u[0] = *(const uint4*)(p);
    f.u[1] = *(const uint4*)(p + 16);
    return f.v;
}

#if USE_ASYNC_LDS
typedef int v4i __attribute__((vector_size(16)));
typedef __attribute__((address_space(1))) v4i gv4i;   // global
typedef __attribute__((address_space(3))) v4i lv4i;   // LDS
__device__ __forceinline__ void async_copy_b128(const void* g, void* l) {
    __builtin_amdgcn_global_load_async_to_lds_b128(
        (gv4i*)(unsigned long long)g,
        (lv4i*)(unsigned)(unsigned long long)l, 0, 0);
}
#endif

// ---------------------------------------------------------------------------
// fp32 -> bf16 conversion
// ---------------------------------------------------------------------------
__global__ __launch_bounds__(256)
void f2bf_kernel(const float* __restrict__ src, unsigned short* __restrict__ dst,
                 int n) {
    int i = blockIdx.x * 256 + threadIdx.x;
    if (i < n) dst[i] = f2bf(src[i]);
}

// ---------------------------------------------------------------------------
// C[4096,1024] = A_bf16[4096,1024] @ W_bf16[1024,1024]^T + bias
// mode 0: f32 out; mode 1: bf16 out (scaled); mode 2: bf16 scattered V^T
//   (V^T layout: [b][h][hd][t] so PV B-fragments are contiguous along s)
// Block = 256 threads (8 waves); tile 128x128, BK=32.
// Wave w: wm = w>>1 owns M strip wm*32 (2 M-tiles); wn = w&1 owns N strip
// wn*64 (4 N-tiles) -> 8 WMMAs per K-step from 6 fragment loads.
// ---------------------------------------------------------------------------
__global__ __launch_bounds__(256)
void gemm_bf16_wmma(const unsigned short* __restrict__ A,
                    const unsigned short* __restrict__ W,
                    const float* __restrict__ bias,
                    float* __restrict__ outF,
                    unsigned short* __restrict__ outH,
                    int mode, float scale) {
#if USE_ASYNC_LDS
    __shared__ unsigned short As[2][128 * 40];
    __shared__ unsigned short Bs[2][128 * 40];
#else
    __shared__ unsigned short As[1][128 * 40];
    __shared__ unsigned short Bs[1][128 * 40];
#endif
    int tid  = threadIdx.x;
    int w    = tid >> 5, lane = tid & 31;
    int wm   = w >> 1,  wn   = w & 1;
    int m0   = blockIdx.y * 128;
    int n0   = blockIdx.x * 128;

    // Tile-fill mapping: 512 chunks of 8 bf16 (16B) per operand, 2 per thread.
    int c0 = tid * 2, c1 = tid * 2 + 1;
    int rA0 = c0 >> 2, sA0 = (c0 & 3) * 8;
    int rA1 = c1 >> 2, sA1 = (c1 & 3) * 8;

    v8f acc[8] = {};   // [i*4+j]: i = M-tile (0..1), j = N-tile (0..3)

#if USE_ASYNC_LDS
    // ---- double-buffered async global->LDS pipeline -----------------------
    auto issue = [&](int k0, int p) {
        async_copy_b128(&A[(size_t)(m0 + rA0) * DD + k0 + sA0], &As[p][rA0 * 40 + sA0]);
        async_copy_b128(&A[(size_t)(m0 + rA1) * DD + k0 + sA1], &As[p][rA1 * 40 + sA1]);
        async_copy_b128(&W[(size_t)(n0 + rA0) * DD + k0 + sA0], &Bs[p][rA0 * 40 + sA0]);
        async_copy_b128(&W[(size_t)(n0 + rA1) * DD + k0 + sA1], &Bs[p][rA1 * 40 + sA1]);
    };
    issue(0, 0);
    int p = 0;
    for (int k0 = 0; k0 < DD; k0 += 32, p ^= 1) {
        if (k0 + 32 < DD) {
            issue(k0 + 32, p ^ 1);
            __builtin_amdgcn_s_wait_asynccnt(4);   // current tile landed (in-order)
        } else {
            __builtin_amdgcn_s_wait_asynccnt(0);
        }
        __syncthreads();
        v16bf af0 = frag_from_lds(As[p], wm * 32,      40);
        v16bf af1 = frag_from_lds(As[p], wm * 32 + 16, 40);
        #pragma unroll
        for (int j = 0; j < 4; ++j) {
            v16bf bf = frag_from_lds(Bs[p], wn * 64 + j * 16, 40);
            acc[0 * 4 + j] = __builtin_amdgcn_wmma_f32_16x16x32_bf16(
                false, af0, false, bf, (short)0, acc[0 * 4 + j], false, false);
            acc[1 * 4 + j] = __builtin_amdgcn_wmma_f32_16x16x32_bf16(
                false, af1, false, bf, (short)0, acc[1 * 4 + j], false, false);
        }
        __syncthreads();   // all waves done reading buffer p before reuse
    }
#else
    // ---- register-staged double buffer ------------------------------------
    uint4 ra0, ra1, rb0, rb1;
    auto gload = [&](int k0) {
        ra0 = *(const uint4*)&A[(size_t)(m0 + rA0) * DD + k0 + sA0];
        ra1 = *(const uint4*)&A[(size_t)(m0 + rA1) * DD + k0 + sA1];
        rb0 = *(const uint4*)&W[(size_t)(n0 + rA0) * DD + k0 + sA0];
        rb1 = *(const uint4*)&W[(size_t)(n0 + rA1) * DD + k0 + sA1];
    };
    gload(0);
    for (int k0 = 0; k0 < DD; k0 += 32) {
        *(uint4*)&As[0][rA0 * 40 + sA0] = ra0;
        *(uint4*)&As[0][rA1 * 40 + sA1] = ra1;
        *(uint4*)&Bs[0][rA0 * 40 + sA0] = rb0;
        *(uint4*)&Bs[0][rA1 * 40 + sA1] = rb1;
        __syncthreads();
        if (k0 + 32 < DD) gload(k0 + 32);
        if (k0 + 64 < DD) {   // prefetch two tiles ahead (global_prefetch_b8)
            __builtin_prefetch(&A[(size_t)(m0 + rA0) * DD + k0 + 64 + sA0], 0, 1);
            __builtin_prefetch(&W[(size_t)(n0 + rA0) * DD + k0 + 64 + sA0], 0, 1);
        }
        v16bf af0 = frag_from_lds(As[0], wm * 32,      40);
        v16bf af1 = frag_from_lds(As[0], wm * 32 + 16, 40);
        #pragma unroll
        for (int j = 0; j < 4; ++j) {
            v16bf bf = frag_from_lds(Bs[0], wn * 64 + j * 16, 40);
            acc[0 * 4 + j] = __builtin_amdgcn_wmma_f32_16x16x32_bf16(
                false, af0, false, bf, (short)0, acc[0 * 4 + j], false, false);
            acc[1 * 4 + j] = __builtin_amdgcn_wmma_f32_16x16x32_bf16(
                false, af1, false, bf, (short)0, acc[1 * 4 + j], false, false);
        }
        __syncthreads();
    }
#endif

    int nl = lane & 15;
    int hi = lane >> 4;
    #pragma unroll
    for (int j = 0; j < 4; ++j) {
        int col = n0 + wn * 64 + j * 16 + nl;
        float bval = bias[col];
        #pragma unroll
        for (int i = 0; i < 2; ++i) {
            #pragma unroll
            for (int rr = 0; rr < 8; ++rr) {
                int   m = m0 + wm * 32 + i * 16 + hi * 8 + rr;
                float v = (acc[i * 4 + j][rr] + bval) * scale;
                if (mode == 0) {
                    outF[(size_t)m * DD + col] = v;
                } else if (mode == 1) {
                    outH[(size_t)m * DD + col] = f2bf(v);
                } else {                 // V^T scatter: [b][h][hd][t]
                    int bb = m & 3, t = m >> 2, hh = col >> 6, hd = col & 63;
                    outH[((((size_t)bb * HH + hh) * HD + hd) << 10) + t] = f2bf(v);
                }
            }
        }
    }
}

// ---------------------------------------------------------------------------
// Flash attention. Grid (T/16, B, H); block = 128 (4 waves).
// Q pre-scaled by 1/sqrt(HD). Wave w owns ctx N-strip hd = w*16..w*16+15 and
// softmax rows w*4..w*4+3. Score chunks of 256 columns with online max/sum.
// ---------------------------------------------------------------------------
__global__ __launch_bounds__(128)
void attn_kernel(const unsigned short* __restrict__ Q,
                 const unsigned short* __restrict__ Kb,
                 const unsigned short* __restrict__ Vt,
                 unsigned short* __restrict__ ctx) {
    __shared__ float S[16 * 264];   // 16 rows x 256 cols (+8 pad)
    __shared__ float fac[16];
    __shared__ float invs[16];

    int tid = threadIdx.x, w = tid >> 5, lane = tid & 31;
    int t0 = blockIdx.x * 16, b = blockIdx.y, h = blockIdx.z;

    const unsigned short* qbase = Q + ((size_t)(t0 * BB + b)) * DD + h * HD;
    v16bf aq0 = frag_from_global(qbase,      BB * DD);
    v16bf aq1 = frag_from_global(qbase + 32, BB * DD);

    v8f acc = {};
    float runmax[4], runsum[4];
    #pragma unroll
    for (int i = 0; i < 4; ++i) { runmax[i] = -1e30f; runsum[i] = 0.f; }

    const unsigned short* vbase =
        Vt + (((size_t)b * HH + h) * HD + w * 16) * TT;

    for (int chunk = 0; chunk < 4; ++chunk) {
        int s0 = chunk * 256;
        #pragma unroll
        for (int stq = 0; stq < 4; ++stq) {
            int st = w + stq * 4;
            const unsigned short* kb =
                Kb + ((size_t)((s0 + st * 16) * BB + b)) * DD + h * HD;
            v8f sc = {};
            v16bf bk0 = frag_from_global(kb,      BB * DD);
            sc = __builtin_amdgcn_wmma_f32_16x16x32_bf16(
                false, aq0, false, bk0, (short)0, sc, false, false);
            v16bf bk1 = frag_from_global(kb + 32, BB * DD);
            sc = __builtin_amdgcn_wmma_f32_16x16x32_bf16(
                false, aq1, false, bk1, (short)0, sc, false, false);
            int nl = lane & 15, hi2 = lane >> 4;
            #pragma unroll
            for (int rr = 0; rr < 8; ++rr)
                S[(hi2 * 8 + rr) * 264 + st * 16 + nl] = sc[rr];
        }
        __syncthreads();

        #pragma unroll
        for (int i = 0; i < 4; ++i) {
            int r = w * 4 + i;
            float* row = &S[r * 264];
            float mx = -1e30f;
            for (int j = lane; j < 256; j += 32) mx = fmaxf(mx, row[j]);
            #pragma unroll
            for (int off = 16; off; off >>= 1) mx = fmaxf(mx, __shfl_xor(mx, off));
            float nm = fmaxf(runmax[i], mx);
            float f  = __expf(runmax[i] - nm);
            float s  = 0.f;
            for (int j = lane; j < 256; j += 32) {
                float e = __expf(row[j] - nm);
                row[j] = e;
                s += e;
            }
            #pragma unroll
            for (int off = 16; off; off >>= 1) s += __shfl_xor(s, off);
            runsum[i] = runsum[i] * f + s;
            runmax[i] = nm;
            if (lane == 0) fac[r] = f;
        }
        __syncthreads();

        {
            int hi2 = lane >> 4;
            #pragma unroll
            for (int rr = 0; rr < 8; ++rr) acc[rr] *= fac[hi2 * 8 + rr];
        }
        for (int kt = 0; kt < 8; ++kt) {
            FragU fa;
            int r = lane & 15, half = (lane >> 4) << 3;
            const float* prow = &S[r * 264 + kt * 32 + half];
            #pragma unroll
            for (int jj = 0; jj < 8; ++jj) {
                fa.s[jj]     = f2bf(prow[jj]);
                fa.s[8 + jj] = f2bf(prow[16 + jj]);
            }
            v16bf bv = frag_from_global(vbase + s0 + kt * 32, TT);
            acc = __builtin_amdgcn_wmma_f32_16x16x32_bf16(
                false, fa.v, false, bv, (short)0, acc, false, false);
        }
        __syncthreads();
    }

    if (lane == 0) {
        #pragma unroll
        for (int i = 0; i < 4; ++i) invs[w * 4 + i] = 1.f / runsum[i];
    }
    __syncthreads();

    int nl = lane & 15, hi2 = lane >> 4;
    #pragma unroll
    for (int rr = 0; rr < 8; ++rr) {
        int m = hi2 * 8 + rr;
        float v = acc[rr] * invs[m];
        size_t row = (size_t)(t0 + m) * BB + b;
        ctx[row * DD + h * HD + w * 16 + nl] = f2bf(v);
    }
}

// ---------------------------------------------------------------------------
// LayerNorm(a + res) * g + b ; writes f32 (and optional bf16 copy).
// ---------------------------------------------------------------------------
__global__ __launch_bounds__(256)
void ln_kernel(const float* __restrict__ a, const float* __restrict__ res,
               const float* __restrict__ g, const float* __restrict__ be,
               float* __restrict__ outF, unsigned short* __restrict__ outH) {
    __shared__ float red[16];
    int row = blockIdx.x, tid = threadIdx.x, lane = tid & 31, w = tid >> 5;
    const float* pa = a   + (size_t)row * DD;
    const float* pr = res + (size_t)row * DD;

    float v[4]; float s = 0.f, q = 0.f;
    #pragma unroll
    for (int i = 0; i < 4; ++i) {
        int j = tid + i * 256;
        v[i] = pa[j] + pr[j];
        s += v[i]; q += v[i] * v[i];
    }
    #pragma unroll
    for (int off = 16; off; off >>= 1) {
        s += __shfl_xor(s, off);
        q += __shfl_xor(q, off);
    }
    if (lane == 0) { red[w] = s; red[8 + w] = q; }
    __syncthreads();
    if (tid == 0) {
        float ts = 0.f, tq = 0.f;
        for (int i = 0; i < 8; ++i) { ts += red[i]; tq += red[8 + i]; }
        red[0] = ts; red[8] = tq;
    }
    __syncthreads();
    float mean = red[0] * (1.f / DD);
    float var  = red[8] * (1.f / DD) - mean * mean;
    float rstd = rsqrtf(var + 1e-5f);
    #pragma unroll
    for (int i = 0; i < 4; ++i) {
        int j = tid + i * 256;
        float o = (v[i] - mean) * rstd * g[j] + be[j];
        outF[(size_t)row * DD + j] = o;
        if (outH) outH[(size_t)row * DD + j] = f2bf(o);
    }
}

// ---------------------------------------------------------------------------
extern "C" void kernel_launch(void* const* d_in, const int* in_sizes, int n_in,
                              void* d_out, int out_size, void* d_ws, size_t ws_size,
                              hipStream_t stream) {
    const float* x  = (const float*)d_in[0];
    const float* Wq = (const float*)d_in[1];  const float* bq = (const float*)d_in[2];
    const float* Wk = (const float*)d_in[3];  const float* bk = (const float*)d_in[4];
    const float* Wv = (const float*)d_in[5];  const float* bv = (const float*)d_in[6];
    const float* Wo = (const float*)d_in[7];  const float* bo = (const float*)d_in[8];
    const float* Wl = (const float*)d_in[9];  const float* bl = (const float*)d_in[10];
    const float* g1 = (const float*)d_in[11]; const float* b1 = (const float*)d_in[12];
    const float* g2 = (const float*)d_in[13]; const float* b2 = (const float*)d_in[14];

    size_t off = 0;
    auto alloc = [&](size_t bytes) {
        char* p = (char*)d_ws + off;
        off += (bytes + 255) & ~(size_t)255;
        return p;
    };
    const size_t MD  = (size_t)MM * DD;            // 4096*1024
    unsigned short* xb   = (unsigned short*)alloc(MD * 2);
    unsigned short* Wqb  = (unsigned short*)alloc((size_t)DD * DD * 2);
    unsigned short* Wkb  = (unsigned short*)alloc((size_t)DD * DD * 2);
    unsigned short* Wvb  = (unsigned short*)alloc((size_t)DD * DD * 2);
    unsigned short* Wob  = (unsigned short*)alloc((size_t)DD * DD * 2);
    unsigned short* Wlb  = (unsigned short*)alloc((size_t)DD * DD * 2);
    unsigned short* Qb   = (unsigned short*)alloc(MD * 2);
    unsigned short* Kb   = (unsigned short*)alloc(MD * 2);
    unsigned short* Vt   = (unsigned short*)alloc(MD * 2);
    unsigned short* ctxb = (unsigned short*)alloc(MD * 2);
    float*          aof  = (float*)alloc(MD * 4);  // attn_out, reused for ffn
    float*          hbuf = (float*)alloc(MD * 4);
    unsigned short* hb   = (unsigned short*)alloc(MD * 2);
    (void)ws_size; (void)n_in; (void)in_sizes; (void)out_size;

    const int nX = (int)MD, nW = DD * DD;
    f2bf_kernel<<<(nX + 255) / 256, 256, 0, stream>>>(x,  xb,  nX);
    f2bf_kernel<<<(nW + 255) / 256, 256, 0, stream>>>(Wq, Wqb, nW);
    f2bf_kernel<<<(nW + 255) / 256, 256, 0, stream>>>(Wk, Wkb, nW);
    f2bf_kernel<<<(nW + 255) / 256, 256, 0, stream>>>(Wv, Wvb, nW);
    f2bf_kernel<<<(nW + 255) / 256, 256, 0, stream>>>(Wo, Wob, nW);
    f2bf_kernel<<<(nW + 255) / 256, 256, 0, stream>>>(Wl, Wlb, nW);

    dim3 gg(DD / 128, MM / 128);                   // (8, 32)
    // Q scaled by 1/sqrt(HD)=0.125 (equivalent to scaling scores)
    gemm_bf16_wmma<<<gg, 256, 0, stream>>>(xb, Wqb, bq, nullptr, Qb, 1, 0.125f);
    gemm_bf16_wmma<<<gg, 256, 0, stream>>>(xb, Wkb, bk, nullptr, Kb, 1, 1.0f);
    gemm_bf16_wmma<<<gg, 256, 0, stream>>>(xb, Wvb, bv, nullptr, Vt, 2, 1.0f);

    attn_kernel<<<dim3(TT / 16, BB, HH), 128, 0, stream>>>(Qb, Kb, Vt, ctxb);

    gemm_bf16_wmma<<<gg, 256, 0, stream>>>(ctxb, Wob, bo, aof, nullptr, 0, 1.0f);
    ln_kernel<<<MM, 256, 0, stream>>>(aof, x, g1, b1, hbuf, hb);
    gemm_bf16_wmma<<<gg, 256, 0, stream>>>(hb, Wlb, bl, aof, nullptr, 0, 1.0f);
    ln_kernel<<<MM, 256, 0, stream>>>(aof, hbuf, g2, b2, (float*)d_out, nullptr);
}
